// deformable_LKA_55164559949921
// MI455X (gfx1250) — matrix operand order repaired and checked by reading, not compiled, not run on hardware
//
#include <hip/hip_runtime.h>
#include <hip/hip_bf16.h>

typedef _Float16 v16h __attribute__((ext_vector_type(16)));
typedef _Float16 v8h  __attribute__((ext_vector_type(8)));
typedef _Float16 v4h  __attribute__((ext_vector_type(4)));
typedef float    v8f  __attribute__((ext_vector_type(8)));

#define BB   8
#define CC   64
#define HH   128
#define WW   128
#define HWW  (HH * WW)
#define NTOT (BB * HWW)

// ---------------------------------------------------------------------------
// Pre-pass 1: NCHW f32 -> [n][c] f16 (channel-contiguous, transposed+converted).
// One block per (h, b) row: LDS-transpose so global writes are contiguous b128.
// ---------------------------------------------------------------------------
__global__ __launch_bounds__(256)
void nchw_to_nc16(const float* __restrict__ src, _Float16* __restrict__ dst)
{
  constexpr int LSTR = 72;              // 144B row step: 16B-aligned, bank-clean
  __shared__ _Float16 ld[WW * LSTR];
  const int tid = threadIdx.x;
  const int h = blockIdx.x, b = blockIdx.y;
  const int w  = tid & (WW - 1);
  const int c0 = (tid >> 7) * 32;
  const float* s = src + ((long)b * CC) * HWW + h * WW + w;
  #pragma unroll 8
  for (int i = 0; i < 32; ++i) {        // coalesced across w per c
    int c = c0 + i;
    ld[w * LSTR + c] = (_Float16)s[c * HWW];
  }
  __syncthreads();
  _Float16* o = dst + ((long)(b * HWW + h * WW)) * CC;   // 128*64 halves, contiguous
  #pragma unroll
  for (int j = 0; j < 4; ++j) {
    int off = tid * 32 + j * 8;         // half index in row block
    int ww = off >> 6, cc = off & 63;
    *(v8h*)(o + off) = *(const v8h*)&ld[ww * LSTR + cc]; // 16B aligned both sides
  }
}

// ---------------------------------------------------------------------------
// Pre-pass 2: weights OIHW f32 -> [tap][m][c] f16, m zero-padded to MT*16 rows.
// ---------------------------------------------------------------------------
__global__ __launch_bounds__(256)
void prep_weights(const float* __restrict__ wgt, _Float16* __restrict__ aT,
                  int Cout, int KK, int Ktot, int Mrows, int total)
{
  int idx = blockIdx.x * 256 + threadIdx.x;    // over KK*Mrows*64
  if (idx >= total) return;
  int c = idx & 63;
  int m = (idx >> 6) % Mrows;
  int t = idx / (Mrows * 64);
  float v = (m < Cout) ? wgt[(long)m * Ktot + c * KK + t] : 0.0f;
  aT[idx] = (_Float16)v;
}

// ---------------------------------------------------------------------------
// Implicit-GEMM conv via V_WMMA_F32_16X16X32_F16 (wave32). K is tap-major:
// one 64-wide K round == one kernel tap over all 64 channels.
//
// ONE block owns a 256-N tile and ALL M tiles (MT*16 rows): activations are
// staged once per round and reused by every output channel. 8 waves
// round-robin over MT*16-col jobs; J = 2*MT accumulators per wave.
// Staging per thread per round: 8 x global_load_b128 + 8 x ds_store_b128,
// plus global_prefetch of the NEXT tap's line to overlap with the WMMA block.
// ---------------------------------------------------------------------------
template<int KS, int PAD, int DIL, int MT, bool FUSE_MUL>
__global__ __launch_bounds__(256)
void conv_wmma(const _Float16* __restrict__ inT,   // [n][64] f16
               const _Float16* __restrict__ aT,    // [KK][MT*16][64] f16
               const float* __restrict__ bias, const float* __restrict__ mul,
               float* __restrict__ out, int Cout)
{
  constexpr int KK   = KS * KS;
  constexpr int TN   = 256;
  constexpr int LSTR = 72;            // halves; 144B step (16B-aligned, bank-clean)
  constexpr int J    = MT * 2;        // jobs (16x16 tiles) per wave
  __shared__ _Float16 lb[TN * LSTR];        // activations [n][c]
  __shared__ _Float16 la[MT * 16 * LSTR];   // weights     [m][c]

  const int tid  = threadIdx.x;
  const int lane = tid & 31;
  const int wave = tid >> 5;
  const int g    = lane >> 4;         // half-wave group (CDNA5 wave32 WMMA layout)
  const int m16  = lane & 15;
  const int nBase = blockIdx.x * TN;

  // This thread's staged position (fixed): n = nBase + tid.
  const int n  = nBase + tid;
  const int b  = n >> 14;             // HW == 16384
  const int hw = n & (HWW - 1);
  const int h0 = (hw >> 7) - PAD;
  const int w0 = (hw & (WW - 1)) - PAD;
  const _Float16* inb = inT + (long)b * HWW * CC;

  v8f acc[J] = {};

  #pragma unroll 1
  for (int t = 0; t < KK; ++t) {
    const int ki = t / KS;            // round-uniform (SALU)
    const int kj = t - ki * KS;

    // ---- stage B: this thread's 64 channels at tap t (8 x b128) ---------
    {
      int y = h0 + ki * DIL;
      int x = w0 + kj * DIL;
      bool valid = ((unsigned)y < (unsigned)HH) & ((unsigned)x < (unsigned)WW);
      int yc = min(max(y, 0), HH - 1);
      int xc = min(max(x, 0), WW - 1);
      const v8h* src = (const v8h*)(inb + (long)(yc * WW + xc) * CC);
      v8h* dst = (v8h*)&lb[tid * LSTR];
      if (valid) {
        #pragma unroll
        for (int i = 0; i < 8; ++i) dst[i] = src[i];
      } else {
        v8h z = {};
        #pragma unroll
        for (int i = 0; i < 8; ++i) dst[i] = z;
      }
    }
    // ---- stage A: MT*16 x 64 f16 weights, contiguous (v4h units) --------
    {
      const _Float16* at = aT + (long)t * (MT * 16 * 64);
      #pragma unroll
      for (int j = 0; j < MT; ++j) {
        int f  = tid + 256 * j;       // v4h unit index, < MT*256
        int m  = f >> 4;
        int cb = (f & 15) * 4;
        *(v4h*)&la[m * LSTR + cb] = *(const v4h*)(at + f * 4);
      }
    }
    // ---- prefetch next tap's 128B channel line (overlaps WMMA below) ----
    if (t + 1 < KK) {
      int t2 = t + 1;
      int ki2 = t2 / KS, kj2 = t2 - ki2 * KS;
      int y2 = min(max(h0 + ki2 * DIL, 0), HH - 1);
      int x2 = min(max(w0 + kj2 * DIL, 0), WW - 1);
      const char* p = (const char*)(inb + (long)(y2 * WW + x2) * CC);
      __builtin_prefetch(p,      0, 3);   // global_prefetch_b8 (whole line)
      __builtin_prefetch(p + 64, 0, 3);
    }
    __syncthreads();

    // ---- 2 K-subchunks x J jobs = 2J WMMAs per wave per round -----------
    #pragma unroll
    for (int ks = 0; ks < 2; ++ks) {
      #pragma unroll
      for (int jj = 0; jj < J; ++jj) {
        const int job = wave + 8 * jj;
        const int mt  = job >> 4;     // M tile
        const int col = job & 15;     // N column
        v16h af, bf;
        #pragma unroll
        for (int i = 0; i < 16; ++i) {
          // A 16x32 f16, contiguous k order used here (same as B): group g holds K = g*16 + i, row M = lane&15
          int ka = g * 16 + i;
          af[i] = la[(mt * 16 + m16) * LSTR + ks * 32 + ka];
          // B 32x16 f16, contiguous k order used here (same as A): group g holds K = g*16 + i, column N = lane&15
          bf[i] = lb[(col * 16 + m16) * LSTR + ks * 32 + g * 16 + i];
        }
        acc[jj] = __builtin_amdgcn_wmma_f32_16x16x32_f16(false, af, false, bf,
                                                         (short)0, acc[jj], false, false);
      }
    }
    __syncthreads();
  }

  // ---- epilogue: bias, optional u*attn fuse, store ----------------------
  #pragma unroll
  for (int jj = 0; jj < J; ++jj) {
    const int job = wave + 8 * jj;
    const int mt  = job >> 4;
    const int col = job & 15;
    const int no  = nBase + col * 16 + m16;
    const int ob  = no >> 14;
    const int ohw = no & (HWW - 1);
    #pragma unroll
    for (int r = 0; r < 8; ++r) {
      int m = mt * 16 + g * 8 + r;    // C/D layout: VGPR r holds row M = r + g*8
      if (m < Cout) {
        float v = acc[jj][r] + bias[m];
        long o = ((long)(ob * Cout + m)) * HWW + ohw;
        if (FUSE_MUL) v *= mul[o];
        out[o] = v;
      }
    }
  }
}

// ---------------------------------------------------------------------------
// Deformable depthwise conv, channel-chunked: each thread owns CHUNK=8
// channels of one (b,h,w). Offsets, coordinates and the 4 bilinear weights
// are computed ONCE per tap and reused across the chunk (offset traffic /8,
// coordinate VALU /8). Border handling zeroes the *weights* (4 selects/tap),
// so all gathers are unconditional clamped loads that pipeline freely.
// The sampled map (33.5 MB) fits in the 192 MB L2 -> gathers stay on chip.
// ---------------------------------------------------------------------------
template<int KS, int PAD, int DIL, int CHUNK>
__global__ __launch_bounds__(128)
void deform_dw(const float* __restrict__ x, const float* __restrict__ off,
               const float* __restrict__ dw, float* __restrict__ out)
{
  constexpr int KK = KS * KS;
  const int w  = threadIdx.x;
  const int h  = blockIdx.x;
  const int c0 = blockIdx.y * CHUNK;    // block-uniform -> dw reads are scalar
  const int b  = blockIdx.z;
  const float* xb   = x + ((long)(b * CC + c0)) * HWW;
  const float* offb = off + ((long)b * 2 * KK) * HWW + h * WW + w;
  float acc[CHUNK] = {};
  #pragma unroll 1
  for (int t = 0; t < KK; ++t) {
    int ki = t / KS, kj = t - ki * KS;
    float dy = offb[(2 * t    ) * HWW];
    float dx = offb[(2 * t + 1) * HWW];
    float py = (float)(h - PAD + ki * DIL) + dy;
    float px = (float)(w - PAD + kj * DIL) + dx;
    float y0f = floorf(py), x0f = floorf(px);
    int   y0  = (int)y0f,  x0  = (int)x0f;
    float wy  = py - y0f,  wx  = px - x0f;

    bool y0ok = (unsigned)y0       < (unsigned)HH;
    bool y1ok = (unsigned)(y0 + 1) < (unsigned)HH;
    bool x0ok = (unsigned)x0       < (unsigned)WW;
    bool x1ok = (unsigned)(x0 + 1) < (unsigned)WW;
    int y0c = min(max(y0,     0), HH - 1);
    int y1c = min(max(y0 + 1, 0), HH - 1);
    int x0c = min(max(x0,     0), WW - 1);
    int x1c = min(max(x0 + 1, 0), WW - 1);
    int i00 = y0c * WW + x0c, i01 = y0c * WW + x1c;
    int i10 = y1c * WW + x0c, i11 = y1c * WW + x1c;

    float w00 = (1.f - wy) * (1.f - wx);
    float w01 = (1.f - wy) * wx;
    float w10 = wy * (1.f - wx);
    float w11 = wy * wx;
    w00 = (y0ok & x0ok) ? w00 : 0.0f;   // zero-pad via weights, not values
    w01 = (y0ok & x1ok) ? w01 : 0.0f;
    w10 = (y1ok & x0ok) ? w10 : 0.0f;
    w11 = (y1ok & x1ok) ? w11 : 0.0f;

    #pragma unroll
    for (int j = 0; j < CHUNK; ++j) {
      const float* xc = xb + (long)j * HWW;
      float s = xc[i00] * w00 + xc[i01] * w01 + xc[i10] * w10 + xc[i11] * w11;
      acc[j] += s * dw[(c0 + j) * KK + t];
    }
  }
  #pragma unroll
  for (int j = 0; j < CHUNK; ++j)
    out[((long)(b * CC + c0 + j)) * HWW + h * WW + w] = acc[j];
}

// ---------------------------------------------------------------------------
extern "C" void kernel_launch(void* const* d_in, const int* in_sizes, int n_in,
                              void* d_out, int out_size, void* d_ws, size_t ws_size,
                              hipStream_t stream)
{
  const float* x       = (const float*)d_in[0];
  const float* off0_w  = (const float*)d_in[1];
  const float* off0_b  = (const float*)d_in[2];
  const float* dw0_w   = (const float*)d_in[3];
  const float* off1_w  = (const float*)d_in[4];
  const float* off1_b  = (const float*)d_in[5];
  const float* dw1_w   = (const float*)d_in[6];
  const float* conv1_w = (const float*)d_in[7];
  const float* conv1_b = (const float*)d_in[8];
  float* out = (float*)d_out;

  // ---- workspace carve (~136.2 MB) --------------------------------------
  char* ws = (char*)d_ws;
  float*     attn0  = (float*)ws;                            // 33,554,432 B
  float*     attn1  = (float*)(ws + 33554432);               // 33,554,432 B
  float*     offbuf = (float*)(ws + 67108864);               // 51,380,224 B (98 ch)
  _Float16*  tT     = (_Float16*)(ws + 118489088);           // 16,777,216 B (shared f16 [n][c])
  _Float16*  a0T    = (_Float16*)(ws + 135266304);           //    204,800 B (25*64*64*2)
  _Float16*  a1T    = (_Float16*)(ws + 135471104);           //    702,464 B (49*112*64*2)
  _Float16*  a2T    = (_Float16*)(ws + 136173568);           //      8,192 B

  dim3 cblk(256);
  dim3 tgrid(HH, BB);
  dim3 dgrid(HH, CC / 8, BB);
  dim3 dblk(WW);

  // ---- weight pre-transforms (tiny, f32 -> f16 [tap][m][c]) -------------
  prep_weights<<<dim3((25 * 64 * 64 + 255) / 256),  cblk, 0, stream>>>(
      off0_w, a0T, 50, 25, 1600, 64,  25 * 64 * 64);
  prep_weights<<<dim3((49 * 112 * 64 + 255) / 256), cblk, 0, stream>>>(
      off1_w, a1T, 98, 49, 3136, 112, 49 * 112 * 64);
  prep_weights<<<dim3((1 * 64 * 64 + 255) / 256),   cblk, 0, stream>>>(
      conv1_w, a2T, 64, 1, 64, 64, 1 * 64 * 64);

  // ---- stage 0: conv5x5 offsets -> deformable depthwise 5x5 -------------
  nchw_to_nc16<<<tgrid, cblk, 0, stream>>>(x, tT);
  conv_wmma<5, 2, 1, 4, false><<<dim3(NTOT / 256), cblk, 0, stream>>>(
      tT, a0T, off0_b, nullptr, offbuf, 50);
  deform_dw<5, 2, 1, 8><<<dgrid, dblk, 0, stream>>>(x, offbuf, dw0_w, attn0);

  // ---- stage 1: conv7x7 dil=3 offsets -> deformable depthwise 7x7 -------
  nchw_to_nc16<<<tgrid, cblk, 0, stream>>>(attn0, tT);
  conv_wmma<7, 9, 3, 7, false><<<dim3(NTOT / 256), cblk, 0, stream>>>(
      tT, a1T, off1_b, nullptr, offbuf, 98);
  deform_dw<7, 9, 3, 8><<<dgrid, dblk, 0, stream>>>(attn0, offbuf, dw1_w, attn1);

  // ---- 1x1 conv (64x64 GEMM) + bias, fused with out = u * attn ----------
  nchw_to_nc16<<<tgrid, cblk, 0, stream>>>(attn1, tT);
  conv_wmma<1, 0, 1, 4, true><<<dim3(NTOT / 256), cblk, 0, stream>>>(
      tT, a2T, conv1_b, x, out, 64);
}
